// NeuralPCFG_91044716741117
// MI455X (gfx1250) — matrix operand order repaired and checked by reading, not compiled, not run on hardware
//
#include <hip/hip_runtime.h>
#include <hip/hip_bf16.h>
#include <cstdint>
#include <cstddef>

#define NEGF (-1e9f)
#define CB 32    // batch
#define CN 30    // max sentence length
#define CNT 45   // nonterminals
#define CPT 45   // preterminals
#define CS 90    // NT+PT
#define CD 256   // hidden dim
#define CV 10000 // vocab
#define EK 8192  // S*S=8100 padded to 256 WMMA K-steps (score GEMM)
#define KSPLIT 4 // K-split of the score GEMM (64 WMMA steps per split)
#define EROWS 960 // E rows allocated (>= 15 groups * 4 tiles * 16)
#define VLD 10016 // vocab padded to multiple of 16
#define RPAD 8112 // S*S padded to multiple of 16 (rule rows)

typedef __attribute__((ext_vector_type(16))) _Float16 v16h;
typedef __attribute__((ext_vector_type(8)))  float    v8f;

__device__ __forceinline__ size_t chart_off(int w, int b, int i) {
  return (((size_t)w * CB + b) * CN + i) * CS;
}

union FragU { v16h v; int4 q[2]; };

// A-matrix 16x32 f16 fragment (row-major source, stride ld halves).
// Lane l holds row m = l%16; halves[0..7] = K (l/16)*8..+7, halves[8..15] = 16+(l/16)*8..+7.
__device__ __forceinline__ v16h frag_a(const _Float16* base, int ld) {
  int lane = threadIdx.x & 31;
  const _Float16* p = base + (size_t)(lane & 15) * ld + ((lane >> 4) * 8);
  FragU u;
  u.q[0] = *reinterpret_cast<const int4*>(p);
  u.q[1] = *reinterpret_cast<const int4*>(p + 16);
  return u.v;
}

// B-matrix 32x16 f16 fragment from a row-major B^T buffer (rows = N cols, stride ld halves).
// Lane l holds col n = l%16; halves[0..7] = K (l/16)*8..+7, halves[8..15] = 16+(l/16)*8..+7.
__device__ __forceinline__ v16h frag_b(const _Float16* base, int ld) {
  int lane = threadIdx.x & 31;
  const _Float16* p = base + (size_t)(lane & 15) * ld + ((lane >> 4) * 8);
  FragU u;
  u.q[0] = *reinterpret_cast<const int4*>(p);
  u.q[1] = *reinterpret_cast<const int4*>(p + 16);
  return u.v;
}

__global__ void fill_neg_k(float* p, long n) {
  long idx = blockIdx.x * 256L + threadIdx.x;
  if (idx < n) p[idx] = NEGF;
}

__global__ void cvt16_k(const float* src, _Float16* dst, int rows, int cols, long total) {
  long idx = blockIdx.x * 256L + threadIdx.x;
  if (idx >= total) return;
  long r = idx / cols;
  dst[idx] = (r < rows) ? (_Float16)src[idx] : (_Float16)0.0f;
}

// out[m, n] = sum_k A16[m,k]*Bt16[n,k] + bias[n]   (M tiles x, N tiles y; 1 wave/block)
__global__ void wmma_gemm_bias_k(const _Float16* A, const _Float16* Bt, const float* bias,
                                 float* outp, int Nvalid, int Mvalid, int ldOut, int K) {
  int m0 = blockIdx.x * 16, n0 = blockIdx.y * 16;
  v8f c = {0.f,0.f,0.f,0.f,0.f,0.f,0.f,0.f};
  for (int kk = 0; kk < K; kk += 32) {
    v16h a  = frag_a(A  + (size_t)m0 * K + kk, K);
    v16h bb = frag_b(Bt + (size_t)n0 * K + kk, K);
    c = __builtin_amdgcn_wmma_f32_16x16x32_f16(false, a, false, bb, (short)0, c, false, false);
  }
  int lane = threadIdx.x & 31;
  int n = n0 + (lane & 15);
  int mb = m0 + ((lane >> 4) * 8);
  for (int j = 0; j < 8; ++j) {
    int m = mb + j;
    if (m < Mvalid && n < Nvalid) outp[(size_t)m * ldOut + n] = c[j] + bias[n];
  }
}

// Per-row max + logsumexp. mode 0: out=-log(sumexp) (rules rmax_a), rowmaxOut=max. mode 1: out=max+log(sumexp).
__global__ void row_reduce_k(const float* X, int ld, int n, float* rowmaxOut, float* outv, int mode) {
  int r = blockIdx.x, tid = threadIdx.x;
  __shared__ float red[256];
  const float* row = X + (size_t)r * ld;
  float m = -3.0e38f;
  for (int j = tid; j < n; j += 256) m = fmaxf(m, row[j]);
  red[tid] = m; __syncthreads();
  for (int o = 128; o > 0; o >>= 1) { if (tid < o) red[tid] = fmaxf(red[tid], red[tid + o]); __syncthreads(); }
  m = red[0]; __syncthreads();
  float s = 0.f;
  for (int j = tid; j < n; j += 256) s += __expf(row[j] - m);
  red[tid] = s; __syncthreads();
  for (int o = 128; o > 0; o >>= 1) { if (tid < o) red[tid] += red[tid + o]; __syncthreads(); }
  if (tid == 0) {
    if (rowmaxOut) rowmaxOut[r] = m;
    outv[r] = mode ? (m + __logf(red[0])) : (-__logf(red[0]));
  }
}

__global__ void er_cvt_k(const float* RL, const float* rowmax, _Float16* Er) {
  int idx = blockIdx.x * 256 + threadIdx.x;
  if (idx >= 48 * EK) return;
  int a = idx / EK, k = idx % EK;
  float e = 0.f;
  if (a < CNT && k < CS * CS) e = __expf(RL[(size_t)a * (CS * CS) + k] - rowmax[a]);
  Er[idx] = (_Float16)e;
}

// out[m,o] = (relu?)(X[m,:]@W[o,:] + bias[o]) (+ res[m,o])
__global__ void lin_k(const float* X, const float* W, const float* bias, const float* res,
                      float* out, int Din, int Dout, int relu) {
  int o = blockIdx.x * blockDim.x + threadIdx.x;
  int m = blockIdx.y;
  if (o >= Dout) return;
  const float* xr = X + (size_t)m * Din;
  const float* wr = W + (size_t)o * Din;
  float acc = bias[o];
  for (int d = 0; d < Din; ++d) acc += xr[d] * wr[d];
  if (relu) acc = fmaxf(acc, 0.f);
  if (res) acc += res[(size_t)m * Dout + o];
  out[(size_t)m * Dout + o] = acc;
}

__global__ void softmax45_k(const float* in, float* outv) {
  __shared__ float red[64];
  int tid = threadIdx.x;
  float v = (tid < CNT) ? in[tid] : NEGF;
  red[tid] = v; __syncthreads();
  for (int o = 32; o > 0; o >>= 1) { if (tid < o) red[tid] = fmaxf(red[tid], red[tid + o]); __syncthreads(); }
  float m = red[0]; __syncthreads();
  float e = (tid < CNT) ? __expf(v - m) : 0.f;
  red[tid] = e; __syncthreads();
  for (int o = 32; o > 0; o >>= 1) { if (tid < o) red[tid] += red[tid + o]; __syncthreads(); }
  if (tid < CNT) outv[tid] = v - (m + __logf(red[0]));
}

__global__ void gather_terms_k(const float* VL, const float* logZ, const int* x, float* chart) {
  int idx = blockIdx.x * 256 + threadIdx.x;
  if (idx >= CB * CN * CPT) return;
  int pt = idx % CPT;
  int n = (idx / CPT) % CN;
  int b = idx / (CPT * CN);
  int tok = x[b * CN + n];
  chart[chart_off(1, b, n) + CNT + pt] = VL[(size_t)pt * VLD + tok] - logZ[pt];
}

// One workgroup per (b,i): outer-product GEMM over splits, Yf, ymax, exp-row E.
__global__ __launch_bounds__(128) void k1_span(const float* chart, _Float16* E, float* ymaxOut, int w) {
  int I = CN - w + 1;
  int bi = blockIdx.x;
  int b = bi / I, i = bi % I;
  __shared__ alignas(16) _Float16 Ls[96 * 32];
  __shared__ alignas(16) _Float16 Rt[96 * 32];
  __shared__ float P[96 * 96];       // also used to stage raw chart slices in phase A
  __shared__ float lmax[96], rmax[96];
  __shared__ float red[128];
  int tid = threadIdx.x;

  // Phase A1: coalesced staging of raw L/R chart slices into LDS.
  float* Lraw = P;                 // [w-1][96]
  float* Rraw = P + 96 * 32;       // [w-1][96]
  int tot = (w - 1) * CS;
  for (int idx = tid; idx < tot; idx += 128) {
    int k = idx / CS + 1;          // split width 1..w-1
    int s = idx % CS;              // consecutive lanes -> consecutive s (coalesced)
    Lraw[(k - 1) * 96 + s] = chart[chart_off(k, b, i) + s];
    Rraw[(k - 1) * 96 + s] = chart[chart_off(w - k, b, i + k) + s];
  }
  __syncthreads();

  // Phase A2: per-state max over splits, then shifted exp rows in f16.
  for (int s = tid; s < 96; s += 128) {
    float ml = NEGF, mr = NEGF;
    if (s < CS) {
      for (int k = 0; k < w - 1; ++k) {
        ml = fmaxf(ml, Lraw[k * 96 + s]);
        mr = fmaxf(mr, Rraw[k * 96 + s]);
      }
    }
    lmax[s] = ml; rmax[s] = mr;
    for (int k = 0; k < 32; ++k) {
      float el = 0.f, er = 0.f;
      if (s < CS && k < w - 1) {
        el = __expf(Lraw[k * 96 + s] - ml);
        er = __expf(Rraw[k * 96 + s] - mr);
      }
      Ls[s * 32 + k] = (_Float16)el;
      Rt[s * 32 + k] = (_Float16)er;
    }
  }
  __syncthreads();

  // Phase B: 6x6 WMMA tiles over 96x96 (overwrites P with the product).
  int wv = tid >> 5;
  for (int tt = wv; tt < 36; tt += 4) {
    int mt = tt / 6, ntl = tt % 6;
    v16h a  = frag_a(Ls + mt * 16 * 32, 32);
    v16h bb = frag_b(Rt + ntl * 16 * 32, 32);
    v8f c = {0.f,0.f,0.f,0.f,0.f,0.f,0.f,0.f};
    c = __builtin_amdgcn_wmma_f32_16x16x32_f16(false, a, false, bb, (short)0, c, false, false);
    int lane = tid & 31;
    int n = ntl * 16 + (lane & 15);
    int mb = mt * 16 + ((lane >> 4) * 8);
    for (int j = 0; j < 8; ++j) P[(mb + j) * 96 + n] = c[j];
  }
  __syncthreads();

  // Phase C: Yf = lmax + rmax + log(P+eps); block max.
  float lm = NEGF;
  for (int idx = tid; idx < CS * CS; idx += 128) {
    int s = idx / CS, t = idx % CS;
    float y = lmax[s] + rmax[t] + __logf(P[s * 96 + t] + 1e-38f);
    P[s * 96 + t] = y;
    lm = fmaxf(lm, y);
  }
  red[tid] = lm; __syncthreads();
  for (int o = 64; o > 0; o >>= 1) { if (tid < o) red[tid] = fmaxf(red[tid], red[tid + o]); __syncthreads(); }
  float ym = red[0];
  if (tid == 0) ymaxOut[bi] = ym;

  // Phase D: E row = exp(Yf - ymax) in f16 (zero-padded to EK).
  _Float16* Erow = E + (size_t)bi * EK;
  for (int idx = tid; idx < EK; idx += 128) {
    float e = 0.f;
    if (idx < CS * CS) { int s = idx / CS, t = idx % CS; e = __expf(P[s * 96 + t] - ym); }
    Erow[idx] = (_Float16)e;
  }
}

// K-split partial score GEMM: part[ks][bi][a] += E[bi, ksplit] @ Er[a, ksplit]
// Grid: (mgroups, 3 ntiles, KSPLIT). One wave handles 4 M-tiles per N-tile (B-frag reuse).
__global__ void k2_score(const _Float16* E, const _Float16* Er, float* part, int w) {
  int mg = blockIdx.x * 4;          // first m-tile of group
  int n0 = blockIdx.y * 16;
  int ks = blockIdx.z;
  int k0 = ks * (EK / KSPLIT), k1 = k0 + (EK / KSPLIT);
  const _Float16* Bb = Er + (size_t)n0 * EK;
  const _Float16* A0 = E + (size_t)(mg + 0) * 16 * EK;
  const _Float16* A1 = E + (size_t)(mg + 1) * 16 * EK;
  const _Float16* A2 = E + (size_t)(mg + 2) * 16 * EK;
  const _Float16* A3 = E + (size_t)(mg + 3) * 16 * EK;
  __builtin_prefetch(Bb + k0, 0, 3);
  __builtin_prefetch(A0 + k0, 0, 1);
  v8f c0 = {0.f,0.f,0.f,0.f,0.f,0.f,0.f,0.f};
  v8f c1 = c0, c2 = c0, c3 = c0;
  for (int kk = k0; kk < k1; kk += 32) {
    v16h bb = frag_b(Bb + kk, EK);
    v16h a0 = frag_a(A0 + kk, EK);
    c0 = __builtin_amdgcn_wmma_f32_16x16x32_f16(false, a0, false, bb, (short)0, c0, false, false);
    v16h a1 = frag_a(A1 + kk, EK);
    c1 = __builtin_amdgcn_wmma_f32_16x16x32_f16(false, a1, false, bb, (short)0, c1, false, false);
    v16h a2 = frag_a(A2 + kk, EK);
    c2 = __builtin_amdgcn_wmma_f32_16x16x32_f16(false, a2, false, bb, (short)0, c2, false, false);
    v16h a3 = frag_a(A3 + kk, EK);
    c3 = __builtin_amdgcn_wmma_f32_16x16x32_f16(false, a3, false, bb, (short)0, c3, false, false);
  }
  int lane = threadIdx.x & 31;
  int a = n0 + (lane & 15);
  int mbase = ((lane >> 4) * 8);
  float* pp = part + (size_t)ks * EROWS * 48;
  v8f cs[4] = {c0, c1, c2, c3};
  for (int t = 0; t < 4; ++t) {
    int m0 = (mg + t) * 16;
    for (int j = 0; j < 8; ++j) {
      int bi = m0 + mbase + j;     // bi < EROWS by construction
      pp[(size_t)bi * 48 + a] = cs[t][j];
    }
  }
}

// Deterministic combine over K-splits + chart epilogue.
__global__ void k2_combine(float* chart, const float* part, const float* ymax,
                           const float* rmax_a, int w) {
  int I = CN - w + 1; int BI = CB * I;
  int idx = blockIdx.x * 256 + threadIdx.x;
  if (idx >= BI * 48) return;
  int bi = idx / 48, a = idx % 48;
  if (a >= CNT) return;
  float s = 0.f;
  for (int ks = 0; ks < KSPLIT; ++ks) s += part[((size_t)ks * EROWS + bi) * 48 + a];
  int b = bi / I, i = bi % I;
  chart[chart_off(w, b, i) + a] = ymax[bi] + rmax_a[a] + __logf(s + 1e-38f);
}

__global__ void final_k(const float* chart, const float* roots, const int* lengths, float* out) {
  int b = blockIdx.x, tid = threadIdx.x;
  __shared__ float red[64];
  int len = lengths[b];
  float v = NEGF;
  if (tid < CNT) v = roots[tid] + chart[chart_off(len, b, 0) + tid];
  red[tid] = v; __syncthreads();
  for (int o = 32; o > 0; o >>= 1) { if (tid < o) red[tid] = fmaxf(red[tid], red[tid + o]); __syncthreads(); }
  float m = red[0]; __syncthreads();
  float e = (tid < CNT) ? __expf(v - m) : 0.f;
  red[tid] = e; __syncthreads();
  for (int o = 32; o > 0; o >>= 1) { if (tid < o) red[tid] += red[tid + o]; __syncthreads(); }
  if (tid == 0) out[b] = -(m + __logf(red[0]));
}

extern "C" void kernel_launch(void* const* d_in, const int* in_sizes, int n_in,
                              void* d_out, int out_size, void* d_ws, size_t ws_size,
                              hipStream_t stream) {
  (void)in_sizes; (void)n_in; (void)out_size; (void)ws_size;
  const float* pt_emb     = (const float*)d_in[0];
  const float* nt_emb     = (const float*)d_in[1];
  const float* root_emb   = (const float*)d_in[2];
  const float* rule_W     = (const float*)d_in[3];
  const float* rule_b     = (const float*)d_in[4];
  const float* root_in_W  = (const float*)d_in[5];
  const float* root_in_b  = (const float*)d_in[6];
  const float* root_res_W = (const float*)d_in[7];
  const float* root_res_b = (const float*)d_in[8];
  const float* root_out_W = (const float*)d_in[9];
  const float* root_out_b = (const float*)d_in[10];
  const float* voc_in_W   = (const float*)d_in[11];
  const float* voc_in_b   = (const float*)d_in[12];
  const float* voc_res_W  = (const float*)d_in[13];
  const float* voc_res_b  = (const float*)d_in[14];
  const float* voc_out_W  = (const float*)d_in[15];
  const float* voc_out_b  = (const float*)d_in[16];
  const int*   x          = (const int*)d_in[17];
  const int*   lengths    = (const int*)d_in[18];
  float* out = (float*)d_out;

  char* ws = (char*)d_ws;
  size_t off = 0;
  auto alloc = [&](size_t bytes) { size_t r = off; off += (bytes + 255) & ~(size_t)255; return r; };
  float*     chart   = (float*)(ws + alloc((size_t)31 * CB * CN * CS * 4)); // ~10.7MB
  float*     rulesL  = (float*)(ws + alloc((size_t)CNT * CS * CS * 4));
  float*     vocL    = (float*)(ws + alloc((size_t)CPT * VLD * 4));
  _Float16*  Er      = (_Float16*)(ws + alloc((size_t)48 * EK * 2));
  _Float16*  ruleW16 = (_Float16*)(ws + alloc((size_t)RPAD * CD * 2));
  _Float16*  vocW16  = (_Float16*)(ws + alloc((size_t)VLD * CD * 2));
  _Float16*  nt16    = (_Float16*)(ws + alloc((size_t)48 * CD * 2));
  _Float16*  h16     = (_Float16*)(ws + alloc((size_t)48 * CD * 2));
  float*     hA      = (float*)(ws + alloc((size_t)48 * CD * 4));
  float*     hB      = (float*)(ws + alloc((size_t)48 * CD * 4));
  float*     rootA   = (float*)(ws + alloc(CD * 4));
  float*     rootB   = (float*)(ws + alloc(CD * 4));
  float*     rootsLg = (float*)(ws + alloc(64 * 4));
  float*     roots   = (float*)(ws + alloc(64 * 4));
  float*     rowmax  = (float*)(ws + alloc(64 * 4));
  float*     rmax_a  = (float*)(ws + alloc(64 * 4));
  float*     logZ    = (float*)(ws + alloc(64 * 4));
  float*     ymax    = (float*)(ws + alloc(1024 * 4));
  float*     part    = (float*)(ws + alloc((size_t)KSPLIT * EROWS * 48 * 4)); // ~0.7MB
  _Float16*  E       = (_Float16*)(ws + alloc((size_t)EROWS * EK * 2));       // ~15.7MB

  // 1) chart prefill + f16 weight conversions
  { long n = (long)31 * CB * CN * CS;
    fill_neg_k<<<(unsigned)((n + 255) / 256), 256, 0, stream>>>(chart, n); }
  { long t = 48L * CD;   cvt16_k<<<(unsigned)((t + 255) / 256), 256, 0, stream>>>(nt_emb, nt16, CNT, CD, t); }
  { long t = (long)RPAD * CD; cvt16_k<<<(unsigned)((t + 255) / 256), 256, 0, stream>>>(rule_W, ruleW16, CS * CS, CD, t); }
  { long t = (long)VLD * CD;  cvt16_k<<<(unsigned)((t + 255) / 256), 256, 0, stream>>>(voc_out_W, vocW16, CV, CD, t); }

  // 2) rule logits (WMMA), log-softmax pieces, Er
  wmma_gemm_bias_k<<<dim3(3, RPAD / 16), 32, 0, stream>>>(nt16, ruleW16, rule_b, rulesL, CS * CS, CNT, CS * CS, CD);
  row_reduce_k<<<CNT, 256, 0, stream>>>(rulesL, CS * CS, CS * CS, rowmax, rmax_a, 0);
  { long t = 48L * EK; er_cvt_k<<<(unsigned)((t + 255) / 256), 256, 0, stream>>>(rulesL, rowmax, Er); }

  // 3) root MLP (M=1) -> roots log-softmax
  lin_k<<<dim3(4, 1), 64, 0, stream>>>(root_emb, root_in_W, root_in_b, nullptr, rootA, CD, CD, 0);
  for (int l = 0; l < 2; ++l) {
    lin_k<<<dim3(4, 1), 64, 0, stream>>>(rootA, root_res_W + (size_t)(2 * l) * CD * CD,
                                         root_res_b + (2 * l) * CD, nullptr, rootB, CD, CD, 1);
    lin_k<<<dim3(4, 1), 64, 0, stream>>>(rootB, root_res_W + (size_t)(2 * l + 1) * CD * CD,
                                         root_res_b + (2 * l + 1) * CD, rootA, rootA, CD, CD, 1);
  }
  lin_k<<<dim3(1, 1), 64, 0, stream>>>(rootA, root_out_W, root_out_b, nullptr, rootsLg, CD, CNT, 0);
  softmax45_k<<<1, 64, 0, stream>>>(rootsLg, roots);

  // 4) vocab MLP (M=45) -> vocab logits (WMMA) -> terms into chart width 1
  lin_k<<<dim3(4, CPT), 64, 0, stream>>>(pt_emb, voc_in_W, voc_in_b, nullptr, hA, CD, CD, 0);
  for (int l = 0; l < 2; ++l) {
    lin_k<<<dim3(4, CPT), 64, 0, stream>>>(hA, voc_res_W + (size_t)(2 * l) * CD * CD,
                                           voc_res_b + (2 * l) * CD, nullptr, hB, CD, CD, 1);
    lin_k<<<dim3(4, CPT), 64, 0, stream>>>(hB, voc_res_W + (size_t)(2 * l + 1) * CD * CD,
                                           voc_res_b + (2 * l + 1) * CD, hA, hA, CD, CD, 1);
  }
  { long t = 48L * CD; cvt16_k<<<(unsigned)((t + 255) / 256), 256, 0, stream>>>(hA, h16, CPT, CD, t); }
  wmma_gemm_bias_k<<<dim3(3, VLD / 16), 32, 0, stream>>>(h16, vocW16, voc_out_b, vocL, CV, CPT, VLD, CD);
  row_reduce_k<<<CPT, 256, 0, stream>>>(vocL, VLD, CV, nullptr, logZ, 1);
  gather_terms_k<<<(CB * CN * CPT + 255) / 256, 256, 0, stream>>>(vocL, logZ, x, chart);

  // 5) CKY inside pass: sequential widths; WMMA for both einsums, K-split score GEMM
  for (int w = 2; w <= CN; ++w) {
    int I = CN - w + 1;
    int BI = CB * I;                 // multiple of 32
    int mtiles = BI / 16;
    int groups = (mtiles + 3) / 4;
    k1_span<<<BI, 128, 0, stream>>>(chart, E, ymax, w);
    k2_score<<<dim3(groups, 3, KSPLIT), 32, 0, stream>>>(E, Er, part, w);
    k2_combine<<<(BI * 48 + 255) / 256, 256, 0, stream>>>(chart, part, ymax, rmax_a, w);
  }

  // 6) final -logsumexp per sentence
  final_k<<<CB, 64, 0, stream>>>(chart, roots, lengths, out);
}
